// Decoder_52905407152386
// MI455X (gfx1250) — compile-verified, matrix operand order checked by reading, not run on hardware
//
#include <hip/hip_runtime.h>

// ---------------- problem constants (from reference) ----------------
#define V_    2048
#define D_    512
#define H_    1024
#define E2_   1024
#define B_    32
#define T_    512
#define S_    1024
#define NSTEP 513           // T+1 decode steps
#define CK    16            // k-tiles (of 32) staged per LDS chunk

typedef __attribute__((ext_vector_type(16))) __bf16 v16bf;
typedef __attribute__((ext_vector_type(8)))  float  v8f;

// k-offset inside a 32-wide k-tile for fragment element e of `lane`
// (per CDNA5 ISA 16-bit A/B fragment layout: lanes 0-15 carry K 0..7/16..23,
//  lanes 16-31 carry K 8..15/24..31, packed 2 per VGPR)
__device__ __forceinline__ int kfrag(int lane, int e) {
  int kbase = (lane >> 4) << 3;                // 0 or 8
  return (e < 8) ? (kbase + e) : (16 + kbase + (e - 8));
}

__device__ __forceinline__ unsigned short f2bf(float x) {
  union { float f; unsigned u; } v; v.f = x;
  unsigned r = v.u + 0x7fffu + ((v.u >> 16) & 1u);   // round-to-nearest-even
  return (unsigned short)(r >> 16);
}

__device__ __forceinline__ float sigmoidf_(float x) { return 1.0f / (1.0f + expf(-x)); }

// ---------------- weight packing: fp32 [N,K] (K possibly split across two
// sources) -> bf16 WMMA B-fragments: dst[((ntile*KT+kt)*32+lane)*16+e] -------
__global__ void packW_kernel(const float* __restrict__ src0, int k0,
                             const float* __restrict__ src1, int K, int N,
                             unsigned short* __restrict__ dst) {
  int  KT   = K >> 5;
  long gid  = (long)blockIdx.x * blockDim.x + threadIdx.x;
  long total = (long)(N >> 4) * KT * 32;
  if (gid >= total) return;
  int  lane  = (int)(gid & 31);
  long tmp   = gid >> 5;
  int  kt    = (int)(tmp % KT);
  int  ntile = (int)(tmp / KT);
  int  n     = (ntile << 4) + (lane & 15);
  int  s1    = K - k0;
  unsigned short* p = dst + gid * 16;
#pragma unroll
  for (int e = 0; e < 16; ++e) {
    int kk = (kt << 5) + kfrag(lane, e);
    float v = (kk < k0) ? src0[(long)n * k0 + kk]
                        : src1[(long)n * s1 + (kk - k0)];
    p[e] = f2bf(v);
  }
}

__global__ void zero_kernel(float* __restrict__ p, long n) {
  long i = (long)blockIdx.x * blockDim.x + threadIdx.x;
  if (i < n) p[i] = 0.0f;
}

// ---------------- fused LSTM step: gates = A @ Wcat^T + b, then cell --------
// mode 0: A = [emb[tok_t] (512) | att (1024) | h_prev (1024)]   K=2560
// mode 1: A = [x1 (1024) | h_prev (1024)]                        K=2048
// Each wave owns (mtile, htile) and accumulates i/f/g/o tiles for the SAME
// 16 hidden units -> full LSTM cell applied in registers in the epilogue.
__global__ void lstm_step_kernel(
    int KT, const unsigned short* __restrict__ pW,
    const float* __restrict__ bih, const float* __restrict__ bhh,
    int mode, const int* __restrict__ inputs, int t,
    const float* __restrict__ emb,
    const float* __restrict__ att,
    const float* __restrict__ x1,
    const float* __restrict__ hprev, const float* __restrict__ cprev,
    float* __restrict__ hout, float* __restrict__ cout)
{
  __shared__ alignas(32) unsigned short aT[2 * CK * 32 * 16];  // 32 KB
  int tid  = threadIdx.x;
  int lane = tid & 31;
  int w    = tid >> 5;
  int gw   = blockIdx.x * 8 + w;    // 0..127
  int mtile = gw & 1;               // batch tile (2 x 16 rows)
  int htile = gw >> 1;              // hidden tile 0..63

  v8f acc0 = {}, acc1 = {}, acc2 = {}, acc3 = {};

  int nchunk = KT / CK;
  for (int c = 0; c < nchunk; ++c) {
    // ---- stage A chunk into LDS in A-fragment order (bf16) ----
    for (int item = tid; item < 2 * CK * 32; item += 256) {
      int sl  = item & 31;
      int ktl = (item >> 5) & (CK - 1);
      int mt  = item >> 9;                         // /(32*CK)
      int kt  = c * CK + ktl;
      int m   = (mt << 4) + (sl & 15);
      unsigned short* dstp = &aT[((mt * CK + ktl) * 32 + sl) * 16];
#pragma unroll
      for (int e = 0; e < 16; ++e) {
        int kk = (kt << 5) + kfrag(sl, e);
        float v;
        if (mode == 0) {
          if (kk < D_) {
            int tok = (t == 0) ? 1 : inputs[m * T_ + (t - 1)];   // SOS_ID=1
            v = emb[(long)tok * D_ + kk];
          } else if (kk < D_ + E2_) {
            v = att[m * E2_ + (kk - D_)];
          } else {
            v = hprev[m * H_ + (kk - (D_ + E2_))];
          }
        } else {
          v = (kk < H_) ? x1[m * H_ + kk] : hprev[m * H_ + (kk - H_)];
        }
        dstp[e] = f2bf(v);
      }
    }
    __syncthreads();

    // prefetch next chunk's B fragments into cache (global_prefetch_b8)
    if (c + 1 < nchunk) {
      int ktn = (c + 1) * CK;
#pragma unroll
      for (int g = 0; g < 4; ++g) {
        long ntile = (long)g * 64 + htile;
        __builtin_prefetch((const void*)(pW + ((ntile * KT + ktn) * 32 + lane) * 16), 0, 1);
      }
    }

    // ---- WMMA over the chunk: 4 gate tiles per wave ----
    for (int ktl = 0; ktl < CK; ++ktl) {
      int kt = c * CK + ktl;
      v16bf a = *reinterpret_cast<const v16bf*>(&aT[((mtile * CK + ktl) * 32 + lane) * 16]);
#define GATE(gidx, accv)                                                               \
      { long ntile = (long)(gidx) * 64 + htile;                                        \
        v16bf bfr = *reinterpret_cast<const v16bf*>(pW + ((ntile * KT + kt) * 32 + lane) * 16); \
        accv = __builtin_amdgcn_wmma_f32_16x16x32_bf16(false, a, false, bfr,           \
                                                       (short)0, accv, false, false); }
      GATE(0, acc0) GATE(1, acc1) GATE(2, acc2) GATE(3, acc3)
#undef GATE
    }
    __syncthreads();
  }

  // ---- LSTM cell epilogue, fully in registers per lane ----
#pragma unroll
  for (int r = 0; r < 8; ++r) {
    int m = (mtile << 4) + r + ((lane >> 4) << 3);   // batch row
    int n = (htile << 4) + (lane & 15);              // hidden unit
    float gi = acc0[r] + bih[n]          + bhh[n];
    float gf = acc1[r] + bih[n + H_]     + bhh[n + H_];
    float gg = acc2[r] + bih[n + 2*H_]   + bhh[n + 2*H_];
    float go = acc3[r] + bih[n + 3*H_]   + bhh[n + 3*H_];
    float ii = sigmoidf_(gi), ff = sigmoidf_(gf), oo = sigmoidf_(go);
    float cn = ff * cprev[m * H_ + n] + ii * tanhf(gg);
    float hn = oo * tanhf(cn);
    cout[m * H_ + n] = cn;
    hout[m * H_ + n] = hn;
  }
}

// ---------------- generic skinny GEMM: out = act(A @ W^T + bias) ------------
// A = [a0 (k0 cols) | a1 (K-k0 cols)], row strides k0 / K-k0.
__global__ void gemm_act_kernel(
    int KT, const unsigned short* __restrict__ pW,
    const float* __restrict__ a0, int k0, const float* __restrict__ a1,
    const float* __restrict__ bias, float* __restrict__ out, int ldo, int act)
{
  __shared__ alignas(32) unsigned short aT[2 * CK * 32 * 16];
  int tid = threadIdx.x, lane = tid & 31, w = tid >> 5;
  int gw = blockIdx.x * 8 + w;
  int mtile = gw & 1;
  int ntile = gw >> 1;
  int K  = KT << 5;
  int s1 = K - k0;
  v8f acc = {};

  int nchunk = KT / CK;
  for (int c = 0; c < nchunk; ++c) {
    for (int item = tid; item < 2 * CK * 32; item += 256) {
      int sl  = item & 31;
      int ktl = (item >> 5) & (CK - 1);
      int mt  = item >> 9;
      int kt  = c * CK + ktl;
      int m   = (mt << 4) + (sl & 15);
      unsigned short* dstp = &aT[((mt * CK + ktl) * 32 + sl) * 16];
#pragma unroll
      for (int e = 0; e < 16; ++e) {
        int kk = (kt << 5) + kfrag(sl, e);
        float v = (kk < k0) ? a0[(long)m * k0 + kk]
                            : a1[(long)m * s1 + (kk - k0)];
        dstp[e] = f2bf(v);
      }
    }
    __syncthreads();
    if (c + 1 < nchunk) {
      int ktn = (c + 1) * CK;
      __builtin_prefetch((const void*)(pW + (((long)ntile * KT + ktn) * 32 + lane) * 16), 0, 1);
    }
    for (int ktl = 0; ktl < CK; ++ktl) {
      int kt = c * CK + ktl;
      v16bf a = *reinterpret_cast<const v16bf*>(&aT[((mtile * CK + ktl) * 32 + lane) * 16]);
      v16bf b = *reinterpret_cast<const v16bf*>(pW + (((long)ntile * KT + kt) * 32 + lane) * 16);
      acc = __builtin_amdgcn_wmma_f32_16x16x32_bf16(false, a, false, b, (short)0, acc, false, false);
    }
    __syncthreads();
  }
#pragma unroll
  for (int r = 0; r < 8; ++r) {
    int m = (mtile << 4) + r + ((lane >> 4) << 3);
    int n = (ntile << 4) + (lane & 15);
    float v = acc[r] + bias[n];
    if (act) v = tanhf(v);
    out[(long)m * ldo + n] = v;
  }
}

// ---------------- dot-product attention + softmax, one block per batch ------
__global__ void attention_kernel(const float* __restrict__ enc,
                                 const float* __restrict__ h1,
                                 float* __restrict__ att)
{
  __shared__ float hsh[E2_];
  __shared__ float sc[S_];
  __shared__ float red[256];
  int b = blockIdx.x, tid = threadIdx.x, lane = tid & 31, w = tid >> 5;
  const float* eb = enc + (long)b * S_ * E2_;

  for (int i = tid; i < E2_; i += 256) hsh[i] = h1[b * E2_ + i];
  __syncthreads();

  // scores[s] = h1 . enc[b,s,:]
  for (int s = w; s < S_; s += 8) {
    const float4* row = reinterpret_cast<const float4*>(eb + (long)s * E2_);
    float acc = 0.0f;
#pragma unroll
    for (int p = 0; p < 8; ++p) {
      int idx = p * 32 + lane;            // float4 index, 256 per row
      float4 e = row[idx];
      int hb = idx * 4;
      acc += e.x * hsh[hb] + e.y * hsh[hb + 1] + e.z * hsh[hb + 2] + e.w * hsh[hb + 3];
    }
    for (int off = 16; off; off >>= 1) acc += __shfl_down(acc, off);
    if (lane == 0) sc[s] = acc;
  }
  __syncthreads();

  // softmax over s
  float lm = -3.4e38f;
  for (int i = tid; i < S_; i += 256) lm = fmaxf(lm, sc[i]);
  red[tid] = lm; __syncthreads();
  for (int st = 128; st; st >>= 1) { if (tid < st) red[tid] = fmaxf(red[tid], red[tid + st]); __syncthreads(); }
  float mx = red[0]; __syncthreads();
  float ls = 0.0f;
  for (int i = tid; i < S_; i += 256) { float e = expf(sc[i] - mx); sc[i] = e; ls += e; }
  red[tid] = ls; __syncthreads();
  for (int st = 128; st; st >>= 1) { if (tid < st) red[tid] += red[tid + st]; __syncthreads(); }
  float inv = 1.0f / red[0];
  __syncthreads();

  // att[h] = sum_s w[s]*enc[b,s,h]
  float4 a = {0.f, 0.f, 0.f, 0.f};
  for (int s = 0; s < S_; ++s) {
    float wgt = sc[s];
    float4 e = reinterpret_cast<const float4*>(eb + (long)s * E2_)[tid];
    a.x += wgt * e.x; a.y += wgt * e.y; a.z += wgt * e.z; a.w += wgt * e.w;
  }
  float* ap = att + b * E2_ + tid * 4;
  ap[0] = a.x * inv; ap[1] = a.y * inv; ap[2] = a.z * inv; ap[3] = a.w * inv;
}

// ---------------- per-step NLL: wave b reduces row b; single writer ---------
__global__ void nll_kernel(const float* __restrict__ logits,
                           const int* __restrict__ inputs, int t,
                           float* __restrict__ loss)
{
  __shared__ float part[32];
  int tid = threadIdx.x, lane = tid & 31, b = tid >> 5;   // 1024 threads = 32 waves
  const float* row = logits + b * V_;
  float mx = -3.4e38f;
  for (int j = lane; j < V_; j += 32) mx = fmaxf(mx, row[j]);
  for (int off = 16; off; off >>= 1) mx = fmaxf(mx, __shfl_xor(mx, off));
  float s = 0.0f;
  for (int j = lane; j < V_; j += 32) s += expf(row[j] - mx);
  for (int off = 16; off; off >>= 1) s += __shfl_xor(s, off);
  if (lane == 0) {
    int tok = (t < T_) ? inputs[b * T_ + t] : 2;           // EOS_ID=2
    part[b] = (mx + logf(s)) - row[tok];
  }
  __syncthreads();
  if (tid == 0) {
    float tot = 0.0f;
#pragma unroll
    for (int i = 0; i < 32; ++i) tot += part[i];
    loss[0] += tot;                                        // deterministic single writer
  }
}

__global__ void finalize_kernel(const float* __restrict__ loss, float* __restrict__ out) {
  if (threadIdx.x == 0) out[0] = loss[0] * (1.0f / (float)(B_ * NSTEP));
}

// ---------------- host orchestration ----------------------------------------
extern "C" void kernel_launch(void* const* d_in, const int* in_sizes, int n_in,
                              void* d_out, int out_size, void* d_ws, size_t ws_size,
                              hipStream_t stream)
{
  const int*   inputs = (const int*)d_in[0];
  const float* enc    = (const float*)d_in[1];
  // d_in[2] target_lengths: unused by the reference
  const float* emb    = (const float*)d_in[3];
  const float* Wih0   = (const float*)d_in[4];
  const float* Whh0   = (const float*)d_in[5];
  const float* bih0   = (const float*)d_in[6];
  const float* bhh0   = (const float*)d_in[7];
  const float* Wih1   = (const float*)d_in[8];
  const float* Whh1   = (const float*)d_in[9];
  const float* bih1   = (const float*)d_in[10];
  const float* bhh1   = (const float*)d_in[11];
  const float* W1     = (const float*)d_in[12];
  const float* b1     = (const float*)d_in[13];
  const float* W2     = (const float*)d_in[14];
  const float* b2     = (const float*)d_in[15];

  char* ws = (char*)d_ws;
  size_t off = 0;
  auto take = [&](size_t bytes) -> char* {
    char* p = ws + off;
    off = (off + bytes + 255) & ~(size_t)255;
    return p;
  };

  // packed bf16 weights (B-fragment order)
  unsigned short* pW0 = (unsigned short*)take((size_t)4 * H_ * (D_ + E2_ + H_) * 2); // 4096x2560
  unsigned short* pW1 = (unsigned short*)take((size_t)4 * H_ * (2 * H_) * 2);        // 4096x2048
  unsigned short* pWm = (unsigned short*)take((size_t)H_ * (2 * H_) * 2);            // 1024x2048
  unsigned short* pWo = (unsigned short*)take((size_t)V_ * H_ * 2);                  // 2048x1024

  // fp32 recurrent state (ping-pong) + scratch
  char* stateBase = ws + off;
  float* h0a = (float*)take((size_t)B_ * H_ * 4);
  float* h0b = (float*)take((size_t)B_ * H_ * 4);
  float* c0a = (float*)take((size_t)B_ * H_ * 4);
  float* c0b = (float*)take((size_t)B_ * H_ * 4);
  float* h1a = (float*)take((size_t)B_ * H_ * 4);
  float* h1b = (float*)take((size_t)B_ * H_ * 4);
  float* c1a = (float*)take((size_t)B_ * H_ * 4);
  float* c1b = (float*)take((size_t)B_ * H_ * 4);
  float* att    = (float*)take((size_t)B_ * E2_ * 4);
  float* hid    = (float*)take((size_t)B_ * H_ * 4);
  float* logits = (float*)take((size_t)B_ * V_ * 4);
  float* loss   = (float*)take(256);
  long nzero = (long)(((ws + off) - stateBase) / 4);

  // ---- once per launch: pack weights, zero state ----
  {
    long tot = (long)(4 * H_ / 16) * ((D_ + E2_ + H_) / 32) * 32;
    packW_kernel<<<(int)((tot + 255) / 256), 256, 0, stream>>>(Wih0, D_ + E2_, Whh0, D_ + E2_ + H_, 4 * H_, pW0);
  }
  {
    long tot = (long)(4 * H_ / 16) * ((2 * H_) / 32) * 32;
    packW_kernel<<<(int)((tot + 255) / 256), 256, 0, stream>>>(Wih1, H_, Whh1, 2 * H_, 4 * H_, pW1);
  }
  {
    long tot = (long)(H_ / 16) * ((2 * H_) / 32) * 32;
    packW_kernel<<<(int)((tot + 255) / 256), 256, 0, stream>>>(W1, 2 * H_, W1, 2 * H_, H_, pWm);
  }
  {
    long tot = (long)(V_ / 16) * (H_ / 32) * 32;
    packW_kernel<<<(int)((tot + 255) / 256), 256, 0, stream>>>(W2, H_, W2, H_, V_, pWo);
  }
  zero_kernel<<<(int)((nzero + 255) / 256), 256, 0, stream>>>((float*)stateBase, nzero);

  // ---- 513 recurrent steps ----
  float *h0_in = h0a, *h0_out = h0b, *c0_in = c0a, *c0_out = c0b;
  float *h1_in = h1a, *h1_out = h1b, *c1_in = c1a, *c1_out = c1b;
  for (int t = 0; t < NSTEP; ++t) {
    // LSTM0: A=[emb|att|h0], K=2560 (KT=80)
    lstm_step_kernel<<<16, 256, 0, stream>>>(80, pW0, bih0, bhh0, 0, inputs, t,
                                             emb, att, nullptr,
                                             h0_in, c0_in, h0_out, c0_out);
    // LSTM1: A=[h0_new|h1], K=2048 (KT=64)
    lstm_step_kernel<<<16, 256, 0, stream>>>(64, pW1, bih1, bhh1, 1, nullptr, t,
                                             nullptr, nullptr, h0_out,
                                             h1_in, c1_in, h1_out, c1_out);
    // attention over encoder states
    attention_kernel<<<32, 256, 0, stream>>>(enc, h1_out, att);
    // hid = tanh([h1|att] @ W1^T + b1)   N=1024 -> 16 blocks
    gemm_act_kernel<<<16, 256, 0, stream>>>(64, pWm, h1_out, H_, att, b1, hid, H_, 1);
    // logits = hid @ W2^T + b2           N=2048 -> 32 blocks
    gemm_act_kernel<<<32, 256, 0, stream>>>(32, pWo, hid, H_, hid, b2, logits, V_, 0);
    // accumulate NLL for this step
    nll_kernel<<<1, 1024, 0, stream>>>(logits, inputs, t, loss);

    float* tmp;
    tmp = h0_in; h0_in = h0_out; h0_out = tmp;
    tmp = c0_in; c0_in = c0_out; c0_out = tmp;
    tmp = h1_in; h1_in = h1_out; h1_out = tmp;
    tmp = c1_in; c1_in = c1_out; c1_out = tmp;
  }

  finalize_kernel<<<1, 1, 0, stream>>>(loss, (float*)d_out);
  (void)in_sizes; (void)n_in; (void)out_size; (void)ws_size;
}
